// RQVae_38457137168777
// MI455X (gfx1250) — matrix operand order repaired and checked by reading, not compiled, not run on hardware
//
#include <hip/hip_runtime.h>
#include <hip/hip_bf16.h>
#include <math.h>

// ---------------------------------------------------------------------------
// RQ-VAE forward (losses only) for MI455X / gfx1250, wave32 + WMMA bf16.
//   pre : one-shot f32->bf16 of X / codebooks; weights pre-packed column-major
//         by n into the WMMA k-pair layout (async-copy friendly); codebook
//         0.5*||c||^2 precomputed.
//   enc : Hbf = gelu(X @ W1 + b1) (bf16); Z = Hbf @ W2 + b2 (f32)
//   rq  : 3x nearest-neighbor (WMMA scores + argmax), residual update, loss
//   dec : zq = Z - R on the fly; Hbf = gelu(zq @ dW1 + db1);
//         recon SSE fused into last GEMM (output never stored)
// GEMM: block 128x256, BK=32, 8 waves 2x4, wave tile 64x64 = 16 WMMA accs.
// Staging: GLOBAL_LOAD_ASYNC_TO_LDS_B128 (ASYNCcnt) when available, with
// double-buffered LDS and ONE barrier per K-chunk; sync b128 fallback.
// ---------------------------------------------------------------------------

typedef __attribute__((ext_vector_type(16))) __bf16 v16bf;
typedef __attribute__((ext_vector_type(8)))  float  v8f;

#define DEV static __device__ __forceinline__

DEV unsigned f2bf1(float f) {                 // f32 -> bf16 bits, RNE
    unsigned u = __float_as_uint(f);
    return ((u + 0x7FFFu + ((u >> 16) & 1u)) >> 16) & 0xFFFFu;
}
DEV unsigned packbf(float lo, float hi) { return f2bf1(lo) | (f2bf1(hi) << 16); }
DEV float gelu_exact(float x) { return 0.5f * x * (1.0f + erff(x * 0.70710678118654752f)); }

union BFrag16 { v16bf v; unsigned u[8]; uint4 q[2]; };

// ---- async global->LDS copy (16B per lane), guarded fallback --------------
// Builtin signature (from toolchain diagnostics): pointers to int vector_size(16),
// src in addrspace(1) (__device__), dst in addrspace(3), then imm offset + cpol.
#if __has_builtin(__builtin_amdgcn_global_load_async_to_lds_b128) && \
    __has_builtin(__builtin_amdgcn_s_wait_asynccnt)
#define USE_ASYNC 1
typedef int v4i_vs __attribute__((vector_size(16)));
typedef __attribute__((address_space(1))) v4i_vs* gv4p;
typedef __attribute__((address_space(3))) v4i_vs* lv4p;
#define CP16(g, l)                                                             \
    __builtin_amdgcn_global_load_async_to_lds_b128(                            \
        (gv4p)(const void*)(g), (lv4p)(void*)(l), 0, 0)
#define ASYNC_WAIT() __builtin_amdgcn_s_wait_asynccnt(0)
#else
#define USE_ASYNC 0
#define CP16(g, l) (*(uint4*)(l) = *(const uint4*)(g))
#define ASYNC_WAIT() ((void)0)
#endif

// ============================ one-shot converters ==========================

__global__ __launch_bounds__(256) void cvt_f32_bf16(
    const float* __restrict__ src, unsigned short* __restrict__ dst, int n)
{
    int i = (blockIdx.x * 256 + threadIdx.x) * 8;
    if (i + 8 <= n) {
        float4 a = *(const float4*)(src + i);
        float4 b = *(const float4*)(src + i + 4);
        uint4 o;
        o.x = packbf(a.x, a.y); o.y = packbf(a.z, a.w);
        o.z = packbf(b.x, b.y); o.w = packbf(b.z, b.w);
        *(uint4*)(dst + i) = o;
    }
}

// weights [K][N] f32 -> Bp[n][kp] uint (bf16 pair over adjacent K),
// column-major by n so a column's k-pairs are contiguous (async-friendly).
__global__ __launch_bounds__(256) void pack_w(
    const float* __restrict__ B, unsigned* __restrict__ Bp, int K, int N)
{
    int idx = blockIdx.x * 256 + threadIdx.x;
    int KP = K / 2;
    if (idx < KP * N) {
        int n = idx / KP, kp = idx % KP;
        Bp[idx] = packbf(B[(size_t)(2 * kp) * N + n], B[(size_t)(2 * kp + 1) * N + n]);
    }
}

__global__ __launch_bounds__(256) void halfnorms(
    const float* __restrict__ Cb, float* __restrict__ hn, int Kc)
{
    int e = blockIdx.x * 256 + threadIdx.x;
    if (e < Kc) {
        const float* c = Cb + (size_t)e * 256;
        float s = 0.f;
        for (int i = 0; i < 256; ++i) s += c[i] * c[i];
        hn[e] = 0.5f * s;
    }
}

// ======================= shared GEMM building blocks =======================
// Double-buffered LDS:
//   Al[buf][row 0..127][kpair 0..15]  stride 20 uints
//   Bl[buf][col 0..255][kpair 0..15]  stride 20 uints (fragment-ready)
#define GEMM_PROLOG()                                                          \
    __shared__ unsigned Al[2 * 128 * 20];                                      \
    __shared__ unsigned Bl[2 * 256 * 20];                                      \
    const int tid  = threadIdx.x;                                              \
    const int lane = tid & 31, wid = tid >> 5;                                 \
    const int hi   = lane >> 4, lo = lane & 15;                                \
    const int wm   = wid >> 2, wn = wid & 3;                                   \
    const int bm   = blockIdx.y * 128;                                         \
    const int bn   = blockIdx.x * 256;                                         \
    v8f acc[4][4] = {};                                                        \
    const int arow = tid >> 1, ahalf = tid & 1;

// stage A (bf16 source): 2x 16B per thread
#define GEMM_STAGE_A_BF16(buf, kk)                                             \
    {                                                                          \
        const unsigned* sa = Aw + (size_t)(bm + arow) * KP + ((kk) >> 1) + ahalf * 8; \
        unsigned* da = Al + (buf) * (128 * 20) + arow * 20 + ahalf * 8;        \
        CP16(sa, da); CP16(sa + 4, da + 4);                                    \
    }

// stage B: one column per thread, 4x contiguous 16B copies
#define GEMM_STAGE_B(buf, kk)                                                  \
    {                                                                          \
        const unsigned* sb = Bp + (size_t)(bn + tid) * KP + ((kk) >> 1);       \
        unsigned* db = Bl + (buf) * (256 * 20) + tid * 20;                     \
        CP16(sb, db); CP16(sb + 4, db + 4);                                    \
        CP16(sb + 8, db + 8); CP16(sb + 12, db + 12);                          \
    }

#define GEMM_COMPUTE_CHUNK(buf)                                                \
    {                                                                          \
        const unsigned* Ab = Al + (buf) * (128 * 20);                          \
        const unsigned* Bb = Bl + (buf) * (256 * 20);                          \
        BFrag16 afr[4], bfr[4];                                                \
        _Pragma("unroll")                                                      \
        for (int mi = 0; mi < 4; ++mi) {                                       \
            const int m = wm * 64 + mi * 16 + lo;                              \
            afr[mi].q[0] = *(const uint4*)&Ab[m * 20 + hi * 4];                \
            afr[mi].q[1] = *(const uint4*)&Ab[m * 20 + 8 + hi * 4];            \
        }                                                                      \
        _Pragma("unroll")                                                      \
        for (int ni = 0; ni < 4; ++ni) {                                       \
            const int n = wn * 64 + ni * 16 + lo;                              \
            bfr[ni].q[0] = *(const uint4*)&Bb[n * 20 + hi * 4];                \
            bfr[ni].q[1] = *(const uint4*)&Bb[n * 20 + 8 + hi * 4];            \
        }                                                                      \
        _Pragma("unroll")                                                      \
        for (int mi = 0; mi < 4; ++mi)                                         \
            _Pragma("unroll")                                                  \
            for (int ni = 0; ni < 4; ++ni)                                     \
                acc[mi][ni] = __builtin_amdgcn_wmma_f32_16x16x32_bf16(         \
                    false, afr[mi].v, false, bfr[ni].v, (short)0,              \
                    acc[mi][ni], false, false);                                \
    }

// --------------------- GEMM, bf16 A, bf16-or-f32 output --------------------
template<bool DO_GELU, bool OUT_BF16>
__global__ __launch_bounds__(256)
void gemm_bf16(const unsigned* __restrict__ Aw,   // bf16 A as kpair uints, [M][K/2]
               const unsigned* __restrict__ Bp,   // packed weights [N][K/2]
               const float* __restrict__ bias,
               float* __restrict__ Cf, unsigned short* __restrict__ Ch,
               int M, int N, int K)
{
    GEMM_PROLOG();
    const int KP = K >> 1;
    const int nk = K / 32;

    GEMM_STAGE_A_BF16(0, 0);
    GEMM_STAGE_B(0, 0);
    for (int it = 0; it < nk; ++it) {
        ASYNC_WAIT();
        __syncthreads();
        if (it + 1 < nk) {
            GEMM_STAGE_A_BF16((it + 1) & 1, (it + 1) * 32);
            GEMM_STAGE_B((it + 1) & 1, (it + 1) * 32);
        }
        GEMM_COMPUTE_CHUNK(it & 1);
    }

    #pragma unroll
    for (int mi = 0; mi < 4; ++mi) {
        const int row0 = bm + wm * 64 + mi * 16 + hi * 8;
        #pragma unroll
        for (int ni = 0; ni < 4; ++ni) {
            const int col = bn + wn * 64 + ni * 16 + lo;
            const float bv = bias[col];
            #pragma unroll
            for (int r = 0; r < 8; ++r) {
                float v = acc[mi][ni][r] + bv;
                if (DO_GELU) v = gelu_exact(v);
                if (OUT_BF16) Ch[(size_t)(row0 + r) * N + col] = (unsigned short)f2bf1(v);
                else          Cf[(size_t)(row0 + r) * N + col] = v;
            }
        }
    }
}

// --------------- decoder GEMM 1: A = (Z - R) f32, staged to bf16 -----------
__global__ __launch_bounds__(256)
void gemm_zdiff(const float* __restrict__ Z, const float* __restrict__ R,
                const unsigned* __restrict__ Bp, const float* __restrict__ bias,
                unsigned short* __restrict__ Ch, int M, int N, int K)
{
    GEMM_PROLOG();
    const int KP = K >> 1;
    const int nk = K / 32;

#define STAGE_A_ZDIFF(buf, kk)                                                 \
    {                                                                          \
        const float* zp = Z + (size_t)(bm + arow) * K + (kk) + ahalf * 16;     \
        const float* rp = R + (size_t)(bm + arow) * K + (kk) + ahalf * 16;     \
        unsigned t8[8];                                                        \
        _Pragma("unroll")                                                      \
        for (int j = 0; j < 8; ++j)                                            \
            t8[j] = packbf(zp[2 * j] - rp[2 * j], zp[2 * j + 1] - rp[2 * j + 1]); \
        unsigned* da = Al + (buf) * (128 * 20) + arow * 20 + ahalf * 8;        \
        *(uint4*)&da[0] = make_uint4(t8[0], t8[1], t8[2], t8[3]);              \
        *(uint4*)&da[4] = make_uint4(t8[4], t8[5], t8[6], t8[7]);              \
    }

    GEMM_STAGE_B(0, 0);
    STAGE_A_ZDIFF(0, 0);
    for (int it = 0; it < nk; ++it) {
        ASYNC_WAIT();
        __syncthreads();
        if (it + 1 < nk) {
            GEMM_STAGE_B((it + 1) & 1, (it + 1) * 32);   // async, overlaps A pack
            STAGE_A_ZDIFF((it + 1) & 1, (it + 1) * 32);
        }
        GEMM_COMPUTE_CHUNK(it & 1);
    }
#undef STAGE_A_ZDIFF

    #pragma unroll
    for (int mi = 0; mi < 4; ++mi) {
        const int row0 = bm + wm * 64 + mi * 16 + hi * 8;
        #pragma unroll
        for (int ni = 0; ni < 4; ++ni) {
            const int col = bn + wn * 64 + ni * 16 + lo;
            const float bv = bias[col];
            #pragma unroll
            for (int r = 0; r < 8; ++r)
                Ch[(size_t)(row0 + r) * N + col] =
                    (unsigned short)f2bf1(gelu_exact(acc[mi][ni][r] + bv));
        }
    }
}

// ------------- decoder GEMM 2: bf16 A, fused recon SSE vs X ----------------
__global__ __launch_bounds__(256)
void gemm_sse(const unsigned* __restrict__ Aw, const unsigned* __restrict__ Bp,
              const float* __restrict__ bias, const float* __restrict__ X,
              float* __restrict__ sse_accum, int M, int N, int K)
{
    GEMM_PROLOG();
    __shared__ float red[256];
    const int KP = K >> 1;
    const int nk = K / 32;

    GEMM_STAGE_A_BF16(0, 0);
    GEMM_STAGE_B(0, 0);
    for (int it = 0; it < nk; ++it) {
        ASYNC_WAIT();
        __syncthreads();
        if (it + 1 < nk) {
            GEMM_STAGE_A_BF16((it + 1) & 1, (it + 1) * 32);
            GEMM_STAGE_B((it + 1) & 1, (it + 1) * 32);
        }
        GEMM_COMPUTE_CHUNK(it & 1);
    }

    float s = 0.f;
    #pragma unroll
    for (int mi = 0; mi < 4; ++mi) {
        const int row0 = bm + wm * 64 + mi * 16 + hi * 8;
        #pragma unroll
        for (int ni = 0; ni < 4; ++ni) {
            const int col = bn + wn * 64 + ni * 16 + lo;
            const float bv = bias[col];
            #pragma unroll
            for (int r = 0; r < 8; ++r) {
                float d = acc[mi][ni][r] + bv - X[(size_t)(row0 + r) * N + col];
                s += d * d;
            }
        }
    }
    red[tid] = s;
    __syncthreads();
    for (int off = 128; off > 0; off >>= 1) {
        if (tid < off) red[tid] += red[tid + off];
        __syncthreads();
    }
    if (tid == 0) atomicAdd(sse_accum, red[0]);
}

// ---------------------------------------------------------------------------
// Residual quantization stage. Block = 64 residual rows.
// argmin||r-c||^2 == argmax(r.c - hn[c]); hn precomputed. Codebook chunk
// staged with async b128 copies; residual tile converted once per block.
// ---------------------------------------------------------------------------
__global__ __launch_bounds__(256)
void rq_stage(const float* __restrict__ Rin, float* __restrict__ Rout,
              const unsigned* __restrict__ Cbh,  // bf16 codebook, [Kc][128] uints
              const float* __restrict__ Cb,      // f32 codebook (gather)
              const float* __restrict__ hn,      // 0.5*||c||^2
              int Kc, float* __restrict__ sse_accum)
{
    __shared__ unsigned Al[64 * 132];   // residual: [row][kpair 0..127]
    __shared__ unsigned Cl[32 * 132];   // codebook chunk: [entry][kpair 0..127]
    __shared__ float    scores[64 * 36];
    __shared__ float    hns[32];
    __shared__ float    red[256];
    __shared__ int      bidx[64];

    const int tid  = threadIdx.x;
    const int lane = tid & 31, wid = tid >> 5;
    const int hi   = lane >> 4, lo = lane & 15;
    const int mt   = wid >> 1, nt = wid & 1;
    const int bm   = blockIdx.x * 64;

    {   // cache residual tile 64x256, f32 -> bf16 pairs (once per block)
        const int row = tid >> 2, q = tid & 3;
        const float* ap = Rin + (size_t)(bm + row) * 256 + q * 64;
        #pragma unroll
        for (int g = 0; g < 8; ++g) {
            unsigned t4[4];
            #pragma unroll
            for (int i = 0; i < 4; ++i)
                t4[i] = packbf(ap[g * 8 + 2 * i], ap[g * 8 + 2 * i + 1]);
            *(uint4*)&Al[row * 132 + q * 32 + g * 4] = make_uint4(t4[0], t4[1], t4[2], t4[3]);
        }
    }

    float bestv = -3.0e38f;
    int   bestk = 0;

    for (int e0 = 0; e0 < Kc; e0 += 32) {
        {   // stage 32 codebook rows: 4x contiguous async 16B copies/thread
            const int e = tid >> 3, oct = tid & 7;
            const unsigned* sc = Cbh + (size_t)(e0 + e) * 128 + oct * 16;
            unsigned* dc = Cl + e * 132 + oct * 16;
            CP16(sc, dc); CP16(sc + 4, dc + 4);
            CP16(sc + 8, dc + 8); CP16(sc + 12, dc + 12);
        }
        if (tid < 32) hns[tid] = hn[e0 + tid];
        ASYNC_WAIT();
        __syncthreads();

        // scores(64x32) = R(64x256) @ C^T(256x32); 8 K-steps of WMMA per wave
        v8f acc = {};
        #pragma unroll
        for (int s = 0; s < 8; ++s) {
            BFrag16 af, bf;
            const int m = mt * 16 + lo;
            af.q[0] = *(const uint4*)&Al[m * 132 + s * 16 + hi * 4];
            af.q[1] = *(const uint4*)&Al[m * 132 + s * 16 + 8 + hi * 4];
            const int e = nt * 16 + lo;
            bf.q[0] = *(const uint4*)&Cl[e * 132 + s * 16 + hi * 4];
            bf.q[1] = *(const uint4*)&Cl[e * 132 + s * 16 + 8 + hi * 4];
            acc = __builtin_amdgcn_wmma_f32_16x16x32_bf16(
                false, af.v, false, bf.v, (short)0, acc, false, false);
        }
        {
            const int col = nt * 16 + lo;
            #pragma unroll
            for (int r = 0; r < 8; ++r)
                scores[(mt * 16 + hi * 8 + r) * 36 + col] = acc[r];
        }
        __syncthreads();
        if (tid < 64) {     // per-row running argmax
            #pragma unroll 4
            for (int e = 0; e < 32; ++e) {
                float v = scores[tid * 36 + e] - hns[e];
                if (v > bestv) { bestv = v; bestk = e0 + e; }
            }
        }
        __syncthreads();
    }
    if (tid < 64) bidx[tid] = bestk;
    __syncthreads();

    // gather winning code (f32), update residual, accumulate commit SSE
    float s = 0.f;
    {
        const int row = tid >> 2, q = tid & 3;
        const int idx = bidx[row];
        const float* rp = Rin + (size_t)(bm + row) * 256 + q * 64;
        const float* cp = Cb + (size_t)idx * 256 + q * 64;
        float*       op = Rout + (size_t)(bm + row) * 256 + q * 64;
        for (int i = 0; i < 64; ++i) {
            float d = rp[i] - cp[i];
            op[i] = d;
            s += d * d;
        }
    }
    red[tid] = s;
    __syncthreads();
    for (int off = 128; off > 0; off >>= 1) {
        if (tid < off) red[tid] += red[tid + off];
        __syncthreads();
    }
    if (tid == 0) atomicAdd(sse_accum, red[0]);
}

__global__ void zero4(float* p) { if (threadIdx.x < 4) p[threadIdx.x] = 0.f; }

__global__ void finalize4(const float* __restrict__ accum, float* __restrict__ out)
{
    if (threadIdx.x == 0) {
        out[0] = accum[0] * (1.0f / (65536.0f * 1024.0f));           // recon mse
        out[1] = 1.25f * accum[1] * (1.0f / (65536.0f * 256.0f));    // (BETA+GAMMA)*mse
        out[2] = 1.25f * accum[2] * (1.0f / (65536.0f * 256.0f));
        out[3] = 1.25f * accum[3] * (1.0f / (65536.0f * 256.0f));
    }
}

// ---------------------------------------------------------------------------
extern "C" void kernel_launch(void* const* d_in, const int* in_sizes, int n_in,
                              void* d_out, int out_size, void* d_ws, size_t ws_size,
                              hipStream_t stream)
{
    (void)in_sizes; (void)n_in; (void)out_size; (void)ws_size;

    const float* x      = (const float*)d_in[0];
    const float* enc_w1 = (const float*)d_in[1];
    const float* enc_b1 = (const float*)d_in[2];
    const float* enc_w2 = (const float*)d_in[3];
    const float* enc_b2 = (const float*)d_in[4];
    const float* dec_w1 = (const float*)d_in[5];
    const float* dec_b1 = (const float*)d_in[6];
    const float* dec_w2 = (const float*)d_in[7];
    const float* dec_b2 = (const float*)d_in[8];
    const float* cb0    = (const float*)d_in[9];
    const float* cb1    = (const float*)d_in[10];
    const float* cb2    = (const float*)d_in[11];

    const int N = 65536, DIN = 1024, DH = 512, DZ = 256;

    // workspace carve-up (all 16B aligned)
    char* w = (char*)d_ws;
    unsigned short* Xbf = (unsigned short*)w;                 w += (size_t)N * DIN * 2;
    unsigned short* Hbf = (unsigned short*)w;                 w += (size_t)N * DH * 2;
    float*    Z    = (float*)w;                               w += (size_t)N * DZ * 4;
    float*    R    = (float*)w;                               w += (size_t)N * DZ * 4;
    unsigned* W1p  = (unsigned*)w;                            w += (size_t)(DIN / 2) * DH * 4;
    unsigned* W2p  = (unsigned*)w;                            w += (size_t)(DH / 2) * DZ * 4;
    unsigned* dW1p = (unsigned*)w;                            w += (size_t)(DZ / 2) * DH * 4;
    unsigned* dW2p = (unsigned*)w;                            w += (size_t)(DH / 2) * DIN * 4;
    unsigned short* Cb0h = (unsigned short*)w;                w += (size_t)1024 * 256 * 2;
    unsigned short* Cb1h = (unsigned short*)w;                w += (size_t)512 * 256 * 2;
    unsigned short* Cb2h = (unsigned short*)w;                w += (size_t)256 * 256 * 2;
    float* hn0   = (float*)w;                                 w += 1024 * 4;
    float* hn1   = (float*)w;                                 w += 512 * 4;
    float* hn2   = (float*)w;                                 w += 256 * 4;
    float* accum = (float*)w;
    float* out   = (float*)d_out;

    // ---- one-shot precompute ----
    cvt_f32_bf16<<<(N * (size_t)DIN) / 2048, 256, 0, stream>>>(x, Xbf, N * DIN);
    cvt_f32_bf16<<<(1024 * 256) / 2048, 256, 0, stream>>>(cb0, Cb0h, 1024 * 256);
    cvt_f32_bf16<<<(512 * 256) / 2048, 256, 0, stream>>>(cb1, Cb1h, 512 * 256);
    cvt_f32_bf16<<<(256 * 256) / 2048, 256, 0, stream>>>(cb2, Cb2h, 256 * 256);
    pack_w<<<((DIN / 2) * DH + 255) / 256, 256, 0, stream>>>(enc_w1, W1p, DIN, DH);
    pack_w<<<((DH / 2) * DZ + 255) / 256, 256, 0, stream>>>(enc_w2, W2p, DH, DZ);
    pack_w<<<((DZ / 2) * DH + 255) / 256, 256, 0, stream>>>(dec_w1, dW1p, DZ, DH);
    pack_w<<<((DH / 2) * DIN + 255) / 256, 256, 0, stream>>>(dec_w2, dW2p, DH, DIN);
    halfnorms<<<4, 256, 0, stream>>>(cb0, hn0, 1024);
    halfnorms<<<2, 256, 0, stream>>>(cb1, hn1, 512);
    halfnorms<<<1, 256, 0, stream>>>(cb2, hn2, 256);
    zero4<<<1, 32, 0, stream>>>(accum);

    // ---- encoder ----
    gemm_bf16<true, true><<<dim3(DH / 256, N / 128), 256, 0, stream>>>(
        (const unsigned*)Xbf, W1p, enc_b1, nullptr, Hbf, N, DH, DIN);
    gemm_bf16<false, false><<<dim3(DZ / 256, N / 128), 256, 0, stream>>>(
        (const unsigned*)Hbf, W2p, enc_b2, Z, nullptr, N, DZ, DH);

    // ---- residual quantization ----
    rq_stage<<<N / 64, 256, 0, stream>>>(Z, R, (const unsigned*)Cb0h, cb0, hn0, 1024, accum + 1);
    rq_stage<<<N / 64, 256, 0, stream>>>(R, R, (const unsigned*)Cb1h, cb1, hn1,  512, accum + 2);
    rq_stage<<<N / 64, 256, 0, stream>>>(R, R, (const unsigned*)Cb2h, cb2, hn2,  256, accum + 3);

    // ---- decoder (zq = Z - R on the fly), fused recon loss ----
    gemm_zdiff<<<dim3(DH / 256, N / 128), 256, 0, stream>>>(
        Z, R, dW1p, dec_b1, Hbf, N, DH, DZ);
    gemm_sse<<<dim3(DIN / 256, N / 128), 256, 0, stream>>>(
        (const unsigned*)Hbf, dW2p, dec_b2, x, accum, N, DIN, DH);

    finalize4<<<1, 32, 0, stream>>>(accum, out);
}
